// Attention_v2_21603685499610
// MI455X (gfx1250) — matrix operand order repaired and checked by reading, not compiled, not run on hardware
//
#include <hip/hip_runtime.h>

// ---------------------------------------------------------------------------
// MI455X (gfx1250) fused attention pipeline, wave32 + V_WMMA_F32_16X16X32_F16
// + TDM (tensor_load_to_lds) double-buffered K staging + software pipelining.
// ---------------------------------------------------------------------------

typedef __attribute__((ext_vector_type(16))) _Float16 v16h;
typedef __attribute__((ext_vector_type(8)))  float    v8f;
typedef __attribute__((ext_vector_type(4)))  unsigned int v4u;
typedef __attribute__((ext_vector_type(8)))  int      v8i;
typedef __attribute__((ext_vector_type(4)))  int      v4i;

#define B_    4
#define N_    1024
#define R_    5
#define DIM_  1024
#define H_    16
#define DH_   64
#define RD_   320          // R * DH (attention feature dim)
#define NKV_  1056         // 1 null + 1024 keys, padded to multiple of 32
#define ALPHA_ 128.0f
#define SCALE_ (0.0009765625f / 2.23606797749979f)   // dh^-0.5 / alpha / sqrt(R)
#define NEG_BIG_ (-3.0e38f)

#if __has_builtin(__builtin_amdgcn_tensor_load_to_lds) && \
    __has_builtin(__builtin_amdgcn_s_wait_tensorcnt)
#define USE_TDM 1
#else
#define USE_TDM 0
#endif

union HFrag { uint4 u[2]; v16h v; };

// A-matrix 16x32 f16 fragment (per ISA layout): lane row = lane&15,
// elements 0-7 -> K = half*8 + e, elements 8-15 -> K = 16 + half*8 + e.
__device__ inline v16h load_a_frag(const _Float16* base, int stride, int lane) {
  const _Float16* p = base + (size_t)(lane & 15) * stride + ((lane >> 4) << 3);
  HFrag f;
  f.u[0] = *(const uint4*)(p);
  f.u[1] = *(const uint4*)(p + 16);
  return f.v;
}

// B-matrix 32x16 f16 fragment: lane col = lane&15, elements 0-7 -> K = half*8 + e,
// elements 8-15 -> K = 16 + half*8 + e ("stride" is per-column row stride).
__device__ inline v16h load_b_frag(const _Float16* base, int stride, int lane) {
  const _Float16* p = base + (size_t)(lane & 15) * stride + ((lane >> 4) << 3);
  HFrag f;
  f.u[0] = *(const uint4*)(p);
  f.u[1] = *(const uint4*)(p + 16);
  return f.v;
}

__device__ inline v8f wmma16(v16h a, v16h b, v8f c) {
  return __builtin_amdgcn_wmma_f32_16x16x32_f16(false, a, false, b,
                                                (short)0, c, false, false);
}

#if USE_TDM
// Issue a TDM load of a 32-row x 320-f16 K tile (row pitch 320 f16) into LDS.
// D# group0: count=1, lds_addr, 57-bit global addr, type=2 ("image").
// D# group1: data_size=2B, tensor_dim0=320, tensor_dim1=NKV, tile=320x32,
//            dim0_stride=320, dim1_stride=320 (2D: remaining groups zero).
// This toolchain exposes the 6-arg builtin:
//   (uint32x4 g0, int32x8 g1, int32x4, int32x4, int32x8, i32 cpol)
__device__ inline void tdm_stage_k(const _Float16* gsrc, unsigned lds_byte_addr) {
  unsigned long long ga = (unsigned long long)(uintptr_t)gsrc;
  v4u g0 = { 1u,                                        // count=1
             lds_byte_addr,                             // lds_addr [63:32]
             (unsigned)(ga & 0xFFFFFFFFu),              // global_addr lo
             (unsigned)((ga >> 32) & 0x01FFFFFFu) | (2u << 30) }; // hi + type=2
  v8i g1 = { (int)(1u << 16),                           // data_size code 1 (2B)
             (int)((RD_ & 0xFFFFu) << 16),              // tensor_dim0[15:0]@48
             (int)((NKV_ & 0xFFFFu) << 16),             // dim0 hi=0 | tensor_dim1 lo
             (int)(RD_ << 16),                          // dim1 hi=0 | tile_dim0
             32,                                        // tile_dim1=32, tile_dim2=0
             RD_,                                       // tensor_dim0_stride lo
             (int)((RD_ & 0xFFFFu) << 16),              // stride0 hi=0 | stride1 lo
             0 };                                       // stride1 hi
  v4i z4 = {};
  v8i z8 = {};
  __builtin_amdgcn_tensor_load_to_lds(g0, g1, z4, z4, z8, 0);
}
#endif

// ---------------------------------------------------------------------------
// Weight transpose + f32 -> f16 convert:  in[K][Ncol] -> out[Ncol][K]
// ---------------------------------------------------------------------------
__global__ void transpose_cvt(const float* __restrict__ in,
                              _Float16* __restrict__ out, int K, int Ncol) {
  int idx = blockIdx.x * 256 + threadIdx.x;
  if (idx < K * Ncol) {
    int k = idx / Ncol, n = idx % Ncol;
    out[(size_t)n * K + k] = (_Float16)in[idx];
  }
}

// ---------------------------------------------------------------------------
// Input layernorm: writes xn = LN(x)*g (f16) and xc = x (f16) in one pass.
// ---------------------------------------------------------------------------
__global__ __launch_bounds__(256) void ln_in_kernel(const float* __restrict__ x,
                                                    const float* __restrict__ g,
                                                    _Float16* __restrict__ xn,
                                                    _Float16* __restrict__ xc) {
  __shared__ float sm[8];
  const size_t row = blockIdx.x;
  const float* xr = x + row * DIM_;
  int t = threadIdx.x;
  float v[4]; float s = 0.f;
  #pragma unroll
  for (int i = 0; i < 4; ++i) { v[i] = xr[t + 256 * i]; s += v[i]; }
  #pragma unroll
  for (int msk = 16; msk >= 1; msk >>= 1) s += __shfl_xor(s, msk, 32);
  if ((t & 31) == 0) sm[t >> 5] = s;
  __syncthreads();
  float mean = 0.f;
  #pragma unroll
  for (int w = 0; w < 8; ++w) mean += sm[w];
  mean *= (1.0f / DIM_);
  float vs = 0.f;
  #pragma unroll
  for (int i = 0; i < 4; ++i) { float d = v[i] - mean; vs += d * d; }
  #pragma unroll
  for (int msk = 16; msk >= 1; msk >>= 1) vs += __shfl_xor(vs, msk, 32);
  __syncthreads();
  if ((t & 31) == 0) sm[t >> 5] = vs;
  __syncthreads();
  float var = 0.f;
  #pragma unroll
  for (int w = 0; w < 8; ++w) var += sm[w];
  var *= (1.0f / DIM_);
  float rstd = rsqrtf(var + 1e-5f);
  #pragma unroll
  for (int i = 0; i < 4; ++i) {
    int c = t + 256 * i;
    xn[row * DIM_ + c] = (_Float16)((v[i] - mean) * rstd * g[c]);
    xc[row * DIM_ + c] = (_Float16)v[i];
  }
}

// ---------------------------------------------------------------------------
// Output layernorm, in place on f32 buffer.
// ---------------------------------------------------------------------------
__global__ __launch_bounds__(256) void ln_out_kernel(float* __restrict__ y,
                                                     const float* __restrict__ g) {
  __shared__ float sm[8];
  const size_t row = blockIdx.x;
  float* yr = y + row * DIM_;
  int t = threadIdx.x;
  float v[4]; float s = 0.f;
  #pragma unroll
  for (int i = 0; i < 4; ++i) { v[i] = yr[t + 256 * i]; s += v[i]; }
  #pragma unroll
  for (int msk = 16; msk >= 1; msk >>= 1) s += __shfl_xor(s, msk, 32);
  if ((t & 31) == 0) sm[t >> 5] = s;
  __syncthreads();
  float mean = 0.f;
  #pragma unroll
  for (int w = 0; w < 8; ++w) mean += sm[w];
  mean *= (1.0f / DIM_);
  float vs = 0.f;
  #pragma unroll
  for (int i = 0; i < 4; ++i) { float d = v[i] - mean; vs += d * d; }
  #pragma unroll
  for (int msk = 16; msk >= 1; msk >>= 1) vs += __shfl_xor(vs, msk, 32);
  __syncthreads();
  if ((t & 31) == 0) sm[t >> 5] = vs;
  __syncthreads();
  float var = 0.f;
  #pragma unroll
  for (int w = 0; w < 8; ++w) var += sm[w];
  var *= (1.0f / DIM_);
  float rstd = rsqrtf(var + 1e-5f);
  #pragma unroll
  for (int i = 0; i < 4; ++i) {
    int c = t + 256 * i;
    yr[c] = (v[i] - mean) * rstd * g[c];
  }
}

// ---------------------------------------------------------------------------
// 16x64-per-wave WMMA tile core, software-pipelined one k-step deep:
// next iteration's A + 4 B fragments are in flight while current WMMAs issue,
// so waits become partial s_wait_loadcnt instead of loadcnt==0 drains.
// ---------------------------------------------------------------------------
__device__ inline void gemm_tile_16x64(const _Float16* __restrict__ Arow,
                                       const _Float16* __restrict__ Btcols,
                                       int K, int lane, v8f acc[4]) {
  v16h af  = load_a_frag(Arow, K, lane);
  v16h bf0 = load_b_frag(Btcols, K, lane);
  v16h bf1 = load_b_frag(Btcols + (size_t)16 * K, K, lane);
  v16h bf2 = load_b_frag(Btcols + (size_t)32 * K, K, lane);
  v16h bf3 = load_b_frag(Btcols + (size_t)48 * K, K, lane);
  int k0 = 0;
  for (; k0 + 32 < K; k0 += 32) {
    int k1 = k0 + 32;
    v16h afn = load_a_frag(Arow + k1, K, lane);
    v16h bn0 = load_b_frag(Btcols + k1, K, lane);
    v16h bn1 = load_b_frag(Btcols + (size_t)16 * K + k1, K, lane);
    v16h bn2 = load_b_frag(Btcols + (size_t)32 * K + k1, K, lane);
    v16h bn3 = load_b_frag(Btcols + (size_t)48 * K + k1, K, lane);
    acc[0] = wmma16(af, bf0, acc[0]);
    acc[1] = wmma16(af, bf1, acc[1]);
    acc[2] = wmma16(af, bf2, acc[2]);
    acc[3] = wmma16(af, bf3, acc[3]);
    af = afn; bf0 = bn0; bf1 = bn1; bf2 = bn2; bf3 = bn3;
  }
  acc[0] = wmma16(af, bf0, acc[0]);
  acc[1] = wmma16(af, bf1, acc[1]);
  acc[2] = wmma16(af, bf2, acc[2]);
  acc[3] = wmma16(af, bf3, acc[3]);
}

#define MT_ ((B_ * N_ * R_) / 16)   // 1280 row tiles

// Q = LN(x) @ Wq, scaled by SCALE, scattered to q[b][h][i][r*64+dh] (f16).
__global__ __launch_bounds__(256) void gemm_q_kernel(const _Float16* __restrict__ A,
                                                     const _Float16* __restrict__ Bt,
                                                     _Float16* __restrict__ qout) {
  const int K = DIM_;
  int wid = threadIdx.x >> 5, lane = threadIdx.x & 31;
  int tile = blockIdx.x * 8 + wid;
  int mt = tile % MT_, ng = tile / MT_;             // ng in [0,16)
  v8f z = {};
  v8f acc[4]; acc[0] = z; acc[1] = z; acc[2] = z; acc[3] = z;
  gemm_tile_16x64(A + (size_t)mt * 16 * K, Bt + (size_t)ng * 64 * K, K, lane, acc);
  int half = lane >> 4, colb = lane & 15;
  #pragma unroll
  for (int c = 0; c < 4; ++c) {
    int col = ng * 64 + c * 16 + colb;
    int h = col >> 6, dh = col & 63;
    #pragma unroll
    for (int d = 0; d < 8; ++d) {
      int gr = mt * 16 + d + 8 * half;              // (b, i, r) flattened
      int r = gr % R_; int bi = gr / R_;
      int i = bi % N_; int b = bi / N_;
      size_t idx = (((size_t)b * H_ + h) * N_ + i) * RD_ + r * 64 + dh;
      qout[idx] = (_Float16)(acc[c][d] * SCALE_);
    }
  }
}

// KV = x @ Wkv; cols 0..63 -> K[b][1+i][r*64+c], cols 64..127 -> Vt[b][rd][1+i].
__global__ __launch_bounds__(256) void gemm_kv_kernel(const _Float16* __restrict__ A,
                                                      const _Float16* __restrict__ Bt,
                                                      _Float16* __restrict__ kbuf,
                                                      _Float16* __restrict__ vt) {
  const int K = DIM_;
  int wid = threadIdx.x >> 5, lane = threadIdx.x & 31;
  int tile = blockIdx.x * 8 + wid;
  int mt = tile % MT_, ng = tile / MT_;             // ng in [0,2)
  v8f z = {};
  v8f acc[4]; acc[0] = z; acc[1] = z; acc[2] = z; acc[3] = z;
  gemm_tile_16x64(A + (size_t)mt * 16 * K, Bt + (size_t)ng * 64 * K, K, lane, acc);
  int half = lane >> 4, colb = lane & 15;
  #pragma unroll
  for (int c = 0; c < 4; ++c) {
    int col = ng * 64 + c * 16 + colb;              // 0..127
    #pragma unroll
    for (int d = 0; d < 8; ++d) {
      int gr = mt * 16 + d + 8 * half;
      int r = gr % R_; int bi = gr / R_;
      int i = bi % N_; int b = bi / N_;
      _Float16 val = (_Float16)acc[c][d];
      if (col < 64)
        kbuf[((size_t)b * NKV_ + 1 + i) * RD_ + r * 64 + col] = val;
      else
        vt[((size_t)b * RD_ + r * 64 + (col - 64)) * NKV_ + 1 + i] = val;
    }
  }
}

// Out = o @ Wout -> f32 directly into d_out (LN done in place afterwards).
__global__ __launch_bounds__(256) void gemm_out_kernel(const _Float16* __restrict__ A,
                                                       const _Float16* __restrict__ Bt,
                                                       float* __restrict__ out) {
  const int K = DIM_;
  int wid = threadIdx.x >> 5, lane = threadIdx.x & 31;
  int tile = blockIdx.x * 8 + wid;
  int mt = tile % MT_, ng = tile / MT_;
  v8f z = {};
  v8f acc[4]; acc[0] = z; acc[1] = z; acc[2] = z; acc[3] = z;
  gemm_tile_16x64(A + (size_t)mt * 16 * K, Bt + (size_t)ng * 64 * K, K, lane, acc);
  int half = lane >> 4, colb = lane & 15;
  #pragma unroll
  for (int c = 0; c < 4; ++c) {
    int col = ng * 64 + c * 16 + colb;
    #pragma unroll
    for (int d = 0; d < 8; ++d) {
      size_t gr = (size_t)mt * 16 + d + 8 * half;
      out[gr * DIM_ + col] = acc[c][d];
    }
  }
}

// ---------------------------------------------------------------------------
// Fill null-KV column (j=0) and zero the padded key rows j in [1+N, NKV).
// ---------------------------------------------------------------------------
__global__ void kv_fill_kernel(const float* __restrict__ null_kv,
                               _Float16* __restrict__ kbuf,
                               _Float16* __restrict__ vt) {
  int t = blockIdx.x * 256 + threadIdx.x;
  if (t < B_ * RD_) {
    int b = t / RD_, rd = t % RD_, dh = rd & 63;
    kbuf[((size_t)b * NKV_) * RD_ + rd] = (_Float16)null_kv[dh];
    vt[((size_t)b * RD_ + rd) * NKV_] = (_Float16)null_kv[64 + dh];
    for (int j = 1 + N_; j < NKV_; ++j) {
      kbuf[((size_t)b * NKV_ + j) * RD_ + rd] = (_Float16)0.0f;
      vt[((size_t)b * RD_ + rd) * NKV_ + j]   = (_Float16)0.0f;
    }
  }
}

// ---------------------------------------------------------------------------
// Fused flash attention: one wave per (b, h, 16-query tile).
// K tiles (32x320 f16) are DMA'd into double-buffered LDS by the TDM one
// j-step ahead (TENSORcnt-synchronized); QK fragments then come from LDS.
// V fragments stream from global with a one-deep software pipeline.
// ---------------------------------------------------------------------------
__global__ __launch_bounds__(32) void attn_kernel(const _Float16* __restrict__ q,
                                                  const _Float16* __restrict__ kbuf,
                                                  const _Float16* __restrict__ vt,
                                                  const unsigned char* __restrict__ mask,
                                                  _Float16* __restrict__ o) {
  __shared__ __align__(16) _Float16 lds_p[16 * 32];
#if USE_TDM
  __shared__ __align__(16) _Float16 lds_k[2][32 * RD_];
#endif
  int lane = threadIdx.x;
  int itile = blockIdx.x & 63;
  int h = (blockIdx.x >> 6) & 15;
  int b = blockIdx.x >> 10;
  int half = lane >> 4, colb = lane & 15;

  // Q fragments for 16 query rows x 320 features (kept resident).
  v16h qf[10];
  const _Float16* qbase = q + (((size_t)b * H_ + h) * N_ + itile * 16) * RD_;
  #pragma unroll
  for (int c = 0; c < 10; ++c) qf[c] = load_a_frag(qbase + c * 32, RD_, lane);

  v8f zero = {};
  v8f oacc[20];
  #pragma unroll
  for (int t = 0; t < 20; ++t) oacc[t] = zero;
  float m[8], l[8];
  #pragma unroll
  for (int d = 0; d < 8; ++d) { m[d] = NEG_BIG_; l[d] = 0.f; }

  const unsigned char* mrow = mask + (size_t)b * N_;
  const _Float16* kgbase = kbuf + (size_t)b * NKV_ * RD_;
#if USE_TDM
  tdm_stage_k(kgbase, (unsigned)(uintptr_t)&lds_k[0][0]);   // prefetch j0=0
#endif

  for (int j0 = 0; j0 < NKV_; j0 += 32) {
#if USE_TDM
    int buf = (j0 >> 5) & 1;
    if (j0 + 32 < NKV_) {
      tdm_stage_k(kgbase + (size_t)(j0 + 32) * RD_,
                  (unsigned)(uintptr_t)&lds_k[buf ^ 1][0]);
      __builtin_amdgcn_s_wait_tensorcnt(1);   // current tile complete
    } else {
      __builtin_amdgcn_s_wait_tensorcnt(0);
    }
    asm volatile("" ::: "memory");            // keep LDS reads below the wait
    const _Float16* kb = &lds_k[buf][0];
#else
    const _Float16* kb = kgbase + (size_t)j0 * RD_;
#endif
    float zlog[2][8];
    #pragma unroll
    for (int t = 0; t < 2; ++t) {
      const _Float16* kbt = kb + (size_t)(t * 16) * RD_;
      v8f s = zero;
      v16h kf = load_b_frag(kbt, RD_, lane);
      #pragma unroll
      for (int c = 0; c < 10; ++c) {          // one-deep pipelined over chunks
        v16h kfn = kf;
        if (c < 9) kfn = load_b_frag(kbt + (c + 1) * 32, RD_, lane);
        s = wmma16(qf[c], kf, s);
        kf = kfn;
      }
      int j = j0 + t * 16 + colb;             // this lane's key column
      bool ok = (j == 0) || (j <= N_ && mrow[j - 1] != 0);
      #pragma unroll
      for (int d = 0; d < 8; ++d)
        zlog[t][d] = ok ? s[d] * ALPHA_ : NEG_BIG_;
    }
    // ---- online softmax update (per row = d + 8*half) ----
    float p0[8], p1[8];
    #pragma unroll
    for (int d = 0; d < 8; ++d) {
      float cmax = fmaxf(zlog[0][d], zlog[1][d]);
      cmax = fmaxf(cmax, __shfl_xor(cmax, 1, 32));
      cmax = fmaxf(cmax, __shfl_xor(cmax, 2, 32));
      cmax = fmaxf(cmax, __shfl_xor(cmax, 4, 32));
      cmax = fmaxf(cmax, __shfl_xor(cmax, 8, 32));
      float mnew = fmaxf(m[d], cmax);
      float corr = __expf(m[d] - mnew);
      p0[d] = __expf(zlog[0][d] - mnew);
      p1[d] = __expf(zlog[1][d] - mnew);
      float rs = p0[d] + p1[d];
      rs += __shfl_xor(rs, 1, 32);
      rs += __shfl_xor(rs, 2, 32);
      rs += __shfl_xor(rs, 4, 32);
      rs += __shfl_xor(rs, 8, 32);
      l[d] = l[d] * corr + rs;
      m[d] = mnew;
      #pragma unroll
      for (int t = 0; t < 20; ++t) oacc[t][d] *= corr;
    }
    // ---- D-layout -> A-layout transpose of probabilities via LDS ----
    #pragma unroll
    for (int d = 0; d < 8; ++d) {
      lds_p[(d + 8 * half) * 32 + colb]      = (_Float16)p0[d];
      lds_p[(d + 8 * half) * 32 + 16 + colb] = (_Float16)p1[d];
    }
    HFrag af;
    const _Float16* lp = lds_p + colb * 32 + half * 8;
    af.u[0] = *(const uint4*)(lp);
    af.u[1] = *(const uint4*)(lp + 16);
    // ---- AV: attn(16x32) @ V(32x320), one-deep pipelined V fragments ----
    const _Float16* vbase = vt + (size_t)b * RD_ * NKV_ + j0;
    v16h vf = load_b_frag(vbase, NKV_, lane);
    #pragma unroll
    for (int t = 0; t < 20; ++t) {
      v16h vfn = vf;
      if (t < 19) vfn = load_b_frag(vbase + (size_t)((t + 1) * 16) * NKV_, NKV_, lane);
      oacc[t] = wmma16(af.v, vf, oacc[t]);
      vf = vfn;
    }
  }

  // ---- finalize: divide by row sums, scatter to o[b][i][r][h*64+dh] ----
  float inv[8];
  #pragma unroll
  for (int d = 0; d < 8; ++d) inv[d] = 1.0f / l[d];
  int i0 = itile * 16;
  #pragma unroll
  for (int t = 0; t < 20; ++t) {
    int rd = t * 16 + colb;
    int r = rd >> 6, dh = rd & 63;
    #pragma unroll
    for (int d = 0; d < 8; ++d) {
      int i = i0 + d + 8 * half;
      size_t idx = (((size_t)b * N_ + i) * R_ + r) * DIM_ + h * 64 + dh;
      o[idx] = (_Float16)(oacc[t][d] * inv[d]);
    }
  }
}

// ---------------------------------------------------------------------------
// Host-side orchestration
// ---------------------------------------------------------------------------
extern "C" void kernel_launch(void* const* d_in, const int* in_sizes, int n_in,
                              void* d_out, int out_size, void* d_ws, size_t ws_size,
                              hipStream_t stream) {
  (void)in_sizes; (void)n_in; (void)out_size; (void)ws_size;
  const float* x            = (const float*)d_in[0];
  const unsigned char* mask = (const unsigned char*)d_in[1];
  const float* g_in         = (const float*)d_in[2];
  const float* Wq           = (const float*)d_in[3];
  const float* Wkv          = (const float*)d_in[4];
  const float* null_kv      = (const float*)d_in[5];
  const float* Wout         = (const float*)d_in[6];
  const float* g_out        = (const float*)d_in[7];
  float* out = (float*)d_out;

  char* p = (char*)d_ws;
  auto alloc = [&](size_t bytes) {
    char* r = p;
    p += (bytes + 255) & ~(size_t)255;
    return r;
  };
  _Float16* wq_t   = (_Float16*)alloc((size_t)DIM_ * DIM_ * 2);
  _Float16* wkv_t  = (_Float16*)alloc((size_t)DIM_ * 128 * 2);
  _Float16* wout_t = (_Float16*)alloc((size_t)DIM_ * DIM_ * 2);
  _Float16* xn     = (_Float16*)alloc((size_t)B_ * N_ * R_ * DIM_ * 2);
  _Float16* xc     = (_Float16*)alloc((size_t)B_ * N_ * R_ * DIM_ * 2);
  _Float16* qb     = (_Float16*)alloc((size_t)B_ * H_ * N_ * RD_ * 2);
  _Float16* kb     = (_Float16*)alloc((size_t)B_ * NKV_ * RD_ * 2);
  _Float16* vtb    = (_Float16*)alloc((size_t)B_ * RD_ * NKV_ * 2);
  _Float16* ob     = (_Float16*)alloc((size_t)B_ * N_ * R_ * DIM_ * 2);

  // 1) weight transpose + convert
  transpose_cvt<<<4096, 256, 0, stream>>>(Wq,   wq_t,   DIM_, DIM_);
  transpose_cvt<<<512,  256, 0, stream>>>(Wkv,  wkv_t,  DIM_, 128);
  transpose_cvt<<<4096, 256, 0, stream>>>(Wout, wout_t, DIM_, DIM_);
  // 2) input layernorm + f16 conversion of context
  ln_in_kernel<<<B_ * N_ * R_, 256, 0, stream>>>(x, g_in, xn, xc);
  // 3) projections
  gemm_q_kernel<<<(MT_ * 16) / 8, 256, 0, stream>>>(xn, wq_t, qb);
  gemm_kv_kernel<<<(MT_ * 2) / 8, 256, 0, stream>>>(xc, wkv_t, kb, vtb);
  kv_fill_kernel<<<(B_ * RD_ + 255) / 256, 256, 0, stream>>>(null_kv, kb, vtb);
  // 4) fused flash attention
  attn_kernel<<<B_ * H_ * (N_ / 16), 32, 0, stream>>>(qb, kb, vtb, mask, ob);
  // 5) output projection + in-place layernorm
  gemm_out_kernel<<<(MT_ * 16) / 8, 256, 0, stream>>>(ob, wout_t, out);
  ln_out_kernel<<<B_ * N_ * R_, 256, 0, stream>>>(out, g_out);
}
